// Ex4bitLinear_49770081026424
// MI455X (gfx1250) — matrix operand order repaired and checked by reading, not compiled, not run on hardware
//
#include <hip/hip_runtime.h>

// GPTQ 4-bit dequant + GEMM, tuned for MI455X (gfx1250, wave32, WMMA).
// x:[M,4096] fp32, qweight:[512,11008] i32 (8x4b along K), qzeros:[32,1376] i32,
// scales:[32,11008] fp32, out:[M,11008] fp32.  M = 8192.

#define IN_F   4096
#define OUT_F  11008
#define GSIZE  128

#define BM 128
#define BN 128
#define BK 32
#define LPAD 8
#define LSTRIDE (BK + LPAD)   // 40 halves = 80 B row stride (16B aligned)

typedef __attribute__((ext_vector_type(16))) _Float16 v16h;
typedef __attribute__((ext_vector_type(8)))  _Float16 v8h;
typedef __attribute__((ext_vector_type(8)))  float    v8f;

union FragAB {
    v16h v;
    v8h  h[2];
};

__global__ __launch_bounds__(256, 2)
void gptq4_wmma_gemm_gfx1250(const float* __restrict__ x,
                             const int*   __restrict__ qweight,
                             const int*   __restrict__ qzeros,
                             const float* __restrict__ scales,
                             float* __restrict__ out)
{
    __shared__ _Float16 As[BM][LSTRIDE];   // x tile, f16, [m][k]
    __shared__ _Float16 Bs[BN][LSTRIDE];   // dequant W tile, f16, [n][k] (transposed)

    const int tid  = threadIdx.x;
    const int lane = tid & 31;
    const int wave = tid >> 5;
    const int wm   = wave & 3;   // 4 waves along M: 32 rows each
    const int wn   = wave >> 2;  // 2 waves along N: 64 cols each

    const int lm = lane & 15;    // row/col within a 16x16 tile
    const int lg = lane >> 4;    // lane group (K-half selector)

    const int nBlock = blockIdx.x * BN;   // x-major over N -> L2 reuse of x stripe
    const int mBlock = blockIdx.y * BM;

    // Weight-dequant work assignment: one column per thread, two qweight rows.
    const int wcol   = tid & 127;         // 0..127 (column within N tile)
    const int qrBase = tid >> 7;          // 0..1  -> rows {qrBase, qrBase+2}
    const int nGlob  = nBlock + wcol;
    const int zShift = (nGlob & 7) * 4;

    const v8f vzero = {0.f, 0.f, 0.f, 0.f, 0.f, 0.f, 0.f, 0.f};
    v8f acc[2][4];
#pragma unroll
    for (int i = 0; i < 2; ++i)
#pragma unroll
        for (int j = 0; j < 4; ++j)
            acc[i][j] = vzero;

    for (int k0 = 0; k0 < IN_F; k0 += BK) {
        // ---------------- stage x tile (fp32 -> f16) into LDS ----------------
#pragma unroll
        for (int i = 0; i < 4; ++i) {
            const int li = i * 256 + tid;        // 0..1023 float4 slots
            const int r  = li >> 3;              // 0..127
            const int c  = (li & 7) * 4;         // 0,4,...,28
            const float4 f =
                *(const float4*)(x + (size_t)(mBlock + r) * IN_F + k0 + c);
            As[r][c + 0] = (_Float16)f.x;
            As[r][c + 1] = (_Float16)f.y;
            As[r][c + 2] = (_Float16)f.z;
            As[r][c + 3] = (_Float16)f.w;
        }

        // ---------------- dequantize weight tile into LDS --------------------
        // Group index is constant across this K-slice (BK=32 divides GSIZE=128).
        const int   g  = k0 >> 7;
        const float sc = scales[(size_t)g * OUT_F + nGlob];
        const int   zq = qzeros[(size_t)g * (OUT_F / 8) + (nGlob >> 3)];
        const int   z  = ((zq >> zShift) & 0xF) + 1;     // GPTQ stores zero-1
#pragma unroll
        for (int i = 0; i < 2; ++i) {
            const int qr = qrBase + 2 * i;               // 0..3
            const int q  = qweight[(size_t)((k0 >> 3) + qr) * OUT_F + nGlob];
#pragma unroll
            for (int j = 0; j < 8; ++j) {
                const int w = (q >> (4 * j)) & 0xF;
                Bs[wcol][qr * 8 + j] = (_Float16)((float)(w - z) * sc);
            }
        }

        __syncthreads();

        // ---------------- WMMA compute: 2x4 tiles of 16x16 per wave ----------
        FragAB a[2], b[4];
#pragma unroll
        for (int mt = 0; mt < 2; ++mt) {
            const int r = wm * 32 + mt * 16 + lm;
            // ISA 16-bit A 16x32 layout: lane group lg holds K = lg*8..lg*8+7
            // and K = lg*8+16..lg*8+23 (two contiguous 8-half runs).
            a[mt].h[0] = *(const v8h*)&As[r][lg * 8];
            a[mt].h[1] = *(const v8h*)&As[r][lg * 8 + 16];
        }
#pragma unroll
        for (int nt = 0; nt < 4; ++nt) {
            const int c = wn * 64 + nt * 16 + lm;
            // ISA 16-bit B 32x16 layout: lane group lg holds K = lg*8..lg*8+7
            // and K = lg*8+16..lg*8+23 (two contiguous 8-half runs).
            b[nt].h[0] = *(const v8h*)&Bs[c][lg * 8];
            b[nt].h[1] = *(const v8h*)&Bs[c][lg * 8 + 16];
        }

#pragma unroll
        for (int mt = 0; mt < 2; ++mt)
#pragma unroll
            for (int nt = 0; nt < 4; ++nt)
                acc[mt][nt] = __builtin_amdgcn_wmma_f32_16x16x32_f16(
                    /*neg_a=*/false, a[mt].v,
                    /*neg_b=*/false, b[nt].v,
                    /*c_mod=*/(short)0, acc[mt][nt],
                    /*reuse_a=*/false, /*reuse_b=*/false);

        __syncthreads();
    }

    // ---------------- epilogue: C/D 16x16 f32 VGPR layout --------------------
    // VGPR r: lanes 0-15 -> M=r, N=lane; lanes 16-31 -> M=8+r, N=lane-16.
#pragma unroll
    for (int mt = 0; mt < 2; ++mt) {
        const int rowBase = mBlock + wm * 32 + mt * 16 + lg * 8;
#pragma unroll
        for (int nt = 0; nt < 4; ++nt) {
            const int col = nBlock + wn * 64 + nt * 16 + lm;
#pragma unroll
            for (int r = 0; r < 8; ++r) {
                out[(size_t)(rowBase + r) * OUT_F + col] = acc[mt][nt][r];
            }
        }
    }
}

extern "C" void kernel_launch(void* const* d_in, const int* in_sizes, int n_in,
                              void* d_out, int out_size, void* d_ws, size_t ws_size,
                              hipStream_t stream)
{
    const float* x       = (const float*)d_in[0];
    const int*   qweight = (const int*)  d_in[1];
    const int*   qzeros  = (const int*)  d_in[2];
    const float* scales  = (const float*)d_in[3];
    float*       out     = (float*)d_out;

    const int M = in_sizes[0] / IN_F;   // BATCH*SEQ = 8192

    dim3 grid(OUT_F / BN, M / BM);      // (86, 64); x-major over N for L2 reuse of x
    dim3 block(256);
    gptq4_wmma_gemm_gfx1250<<<grid, block, 0, stream>>>(x, qweight, qzeros, scales, out);
}
